// MultiHeadAttention_76759655514825
// MI455X (gfx1250) — compile-verified, matrix operand order checked by reading, not run on hardware
//
#include <hip/hip_runtime.h>

typedef _Float16 f16;
typedef __attribute__((ext_vector_type(16))) _Float16     v16h;
typedef __attribute__((ext_vector_type(8)))  float        v8f;
typedef __attribute__((ext_vector_type(4)))  unsigned int u32x4;

#define B_   2
#define S_   2048
#define D_   1024
#define H_   16
#define DK_  64
#define E3_  3072
#define SCALE_ 0.125f      // 1/sqrt(64)
#define NEG_ -60000.0f

#define LDP_ 72            // padded LDS row pitch (f16): 144B = 9*16B, bank-skewed
#define KT_  16            // K steps of 64 over K=1024

union FragU { v16h v; u32x4 q[2]; };

// A-style fragment (16x32 f16, M x K) from row-major [rows, ld] matrix.
// Per ISA 7.12.2: lane(0-15)=row M, lanes 16-31 same rows with K offset +8.
// B fragments (K x N, column n per lane) use the same loader with r0 = n0,
// reading a matrix stored as [N, K] row-major (i.e. B column-major).
__device__ __forceinline__ v16h load_frag(const f16* base, int ld, int r0, int k0, int lane) {
  const int m = lane & 15;
  const int half = lane >> 4;
  const f16* p = base + (size_t)(r0 + m) * ld + (k0 + half * 8);
  FragU u;
  u.q[0] = *(const u32x4*)(p);        // K = k0 + half*8 + 0..7
  u.q[1] = *(const u32x4*)(p + 16);   // K = k0 + 16 + half*8 + 0..7
  return u.v;
}

__device__ __forceinline__ v8f wmma16(v16h a, v16h b, v8f c) {
  return __builtin_amdgcn_wmma_f32_16x16x32_f16(false, a, false, b, (short)0, c, false, false);
}

// One 16-byte async global->LDS copy (gfx1250 VGLOBAL op, tracked by ASYNCcnt).
__device__ __forceinline__ void async_b128(unsigned lds_byte_addr, const f16* gsrc) {
  asm volatile("global_load_async_to_lds_b128 %0, %1, off"
               :: "v"(lds_byte_addr), "v"(gsrc) : "memory");
}

// ---------------------------------------------------------------- convert
__global__ __launch_bounds__(256) void convert_f32_f16(const float* __restrict__ src,
                                                       f16* __restrict__ dst, int n) {
  int i = blockIdx.x * blockDim.x + threadIdx.x;
  if (i < n) dst[i] = (f16)src[i];
}

// ============================================================ staged GEMM core
// Block: 256 thr / 8 waves; block tile 128(M) x 128(N); wave tile 32x64.
// K loop: 64-wide steps, double-buffered async global->LDS staging.
// Branchless pipeline: next buffer always issued (last tile clamped), so the
// loop body is a single straight-line path and accumulators stay pinned.
#define GEMM_STAGED_BODY(ABASE, BBASE)                                          \
  __shared__ __attribute__((aligned(16))) f16 Abuf[2][128 * LDP_];              \
  __shared__ __attribute__((aligned(16))) f16 Bbuf[2][128 * LDP_];              \
  const int tid  = threadIdx.x;                                                 \
  const int lane = tid & 31;                                                    \
  const int wave = tid >> 5;                                                    \
  const int wm = wave & 3;          /* 4 M strips of 32 rows  */                \
  const int wn = wave >> 2;         /* 2 N strips of 64 cols  */                \
  const int mblock = blockIdx.x * 128;                                          \
  const int nblock = blockIdx.y * 128;                                          \
  v8f cz = {};                                                                  \
  v8f c00=cz,c01=cz,c02=cz,c03=cz,c10=cz,c11=cz,c12=cz,c13=cz;                  \
  auto issue = [&](int buf, int kt) {                                           \
    _Pragma("unroll")                                                           \
    for (int i = 0; i < 4; ++i) {                                               \
      const int chunk = tid + i * 256;     /* 0..1023: 128 rows x 8 chunks */   \
      const int row = chunk >> 3;                                               \
      const int kc  = chunk & 7;                                                \
      const f16* ga = (ABASE) + (size_t)(mblock + row) * D_ + kt * 64 + kc * 8; \
      async_b128((unsigned)(size_t)&Abuf[buf][row * LDP_ + kc * 8], ga);        \
      const f16* gb = (BBASE) + (size_t)(nblock + row) * D_ + kt * 64 + kc * 8; \
      async_b128((unsigned)(size_t)&Bbuf[buf][row * LDP_ + kc * 8], gb);        \
    }                                                                           \
  };                                                                            \
  issue(0, 0);                                                                  \
  for (int kt = 0; kt < KT_; ++kt) {                                            \
    const int cur = kt & 1;                                                     \
    const int ktn = (kt + 1 < KT_) ? kt + 1 : KT_ - 1;                          \
    issue(cur ^ 1, ktn);                                                        \
    /* in-order async completion: <=8 outstanding == just-issued batch only */  \
    asm volatile("s_wait_asynccnt 0x8" ::: "memory");                           \
    __syncthreads();                                                            \
    const f16* As = Abuf[cur];                                                  \
    const f16* Bs = Bbuf[cur];                                                  \
    _Pragma("unroll")                                                           \
    for (int kk = 0; kk < 64; kk += 32) {                                       \
      v16h a0 = load_frag(As, LDP_, wm * 32,      kk, lane);                    \
      v16h a1 = load_frag(As, LDP_, wm * 32 + 16, kk, lane);                    \
      v16h b0 = load_frag(Bs, LDP_, wn * 64 +  0, kk, lane);                    \
      v16h b1 = load_frag(Bs, LDP_, wn * 64 + 16, kk, lane);                    \
      v16h b2 = load_frag(Bs, LDP_, wn * 64 + 32, kk, lane);                    \
      v16h b3 = load_frag(Bs, LDP_, wn * 64 + 48, kk, lane);                    \
      c00 = wmma16(a0, b0, c00); c10 = wmma16(a1, b0, c10);                     \
      c01 = wmma16(a0, b1, c01); c11 = wmma16(a1, b1, c11);                     \
      c02 = wmma16(a0, b2, c02); c12 = wmma16(a1, b2, c12);                     \
      c03 = wmma16(a0, b3, c03); c13 = wmma16(a1, b3, c13);                     \
    }                                                                           \
    __syncthreads();                                                            \
  }                                                                             \
  v8f c[2][4] = {{c00, c01, c02, c03}, {c10, c11, c12, c13}};

// ---------------------------------------------------------------- QKV GEMM
// C[4096,3072] = Xf16[4096,1024] * Wqkv[3072,1024]^T, scattered into
// Q/K/V [B,H,S,DK] f16, Q pre-scaled by 1/sqrt(dk).
__global__ __launch_bounds__(256) void qkv_gemm(const f16* __restrict__ xb,
                                                const f16* __restrict__ wq,
                                                f16* __restrict__ Q,
                                                f16* __restrict__ K,
                                                f16* __restrict__ V) {
  GEMM_STAGED_BODY(xb, wq)

  const int col = lane & 15, half = lane >> 4;
#pragma unroll
  for (int ti = 0; ti < 2; ++ti) {
#pragma unroll
    for (int tj = 0; tj < 4; ++tj) {
      const int m0 = mblock + wm * 32 + ti * 16 + half * 8;
      const int e  = nblock + wn * 64 + tj * 16 + col;
      const int section = e >> 10;                     // 0=q 1=k 2=v
      const int idx = e & 1023;
      const int h = idx >> 6, cc = idx & 63;
      f16* dst = (section == 0) ? Q : ((section == 1) ? K : V);
      const float scl = (section == 0) ? SCALE_ : 1.0f;
#pragma unroll
      for (int r = 0; r < 8; ++r) {
        const int m = m0 + r;
        const int b = m >> 11, s = m & 2047;
        dst[(((size_t)(b * H_ + h)) * S_ + s) * DK_ + cc] = (f16)(c[ti][tj][r] * scl);
      }
    }
  }
}

// ---------------------------------------------------------------- out proj
// out[4096,1024] = ctx_f16[4096,1024] * Wout[1024,1024]^T  (fp32 stores)
__global__ __launch_bounds__(256) void out_gemm(const f16* __restrict__ ctx,
                                                const f16* __restrict__ wo,
                                                float* __restrict__ out) {
  GEMM_STAGED_BODY(ctx, wo)

  const int col = lane & 15, half = lane >> 4;
#pragma unroll
  for (int ti = 0; ti < 2; ++ti) {
#pragma unroll
    for (int tj = 0; tj < 4; ++tj) {
      const int m0 = mblock + wm * 32 + ti * 16 + half * 8;
      const int e  = nblock + wn * 64 + tj * 16 + col;
#pragma unroll
      for (int r = 0; r < 8; ++r)
        out[(size_t)(m0 + r) * D_ + e] = c[ti][tj][r];
    }
  }
}

// ---------------------------------------------------------------- attention
// Block = 4 waves (128 thr): one (b,h) head, 64 q-rows; each wave owns 16.
// Flash-style online softmax over 64-key tiles.
__global__ __launch_bounds__(128) void attention(const f16* __restrict__ Q,
                                                 const f16* __restrict__ Km,
                                                 const f16* __restrict__ Vm,
                                                 const unsigned char* __restrict__ mask,
                                                 f16* __restrict__ ctx) {
  __shared__ __attribute__((aligned(16))) f16 vt[DK_ * 64];      // V^T tile [dk][key]
  __shared__ __attribute__((aligned(16))) f16 pbuf[4 * 16 * 64]; // per-wave P tiles

  const int lane = threadIdx.x & 31;
  const int wave = threadIdx.x >> 5;                 // 0..3
  const int qt = blockIdx.x;                          // 0..31
  const int bh = blockIdx.y;                          // 0..31
  const int b = bh >> 4, h = bh & 15;

  const f16* Qh = Q  + (size_t)bh * S_ * DK_;
  const f16* Kh = Km + (size_t)bh * S_ * DK_;
  const f16* Vh = Vm + (size_t)bh * S_ * DK_;
  const unsigned char* mrow = mask + (size_t)b * S_;

  const int qrow0 = qt * 64 + wave * 16;
  const int col = lane & 15, half = lane >> 4;
  f16* pw = pbuf + wave * (16 * 64);

  // Q fragments are loop-invariant (already scaled by 1/sqrt(dk))
  const v16h aq0 = load_frag(Qh, DK_, qrow0, 0,  lane);
  const v16h aq1 = load_frag(Qh, DK_, qrow0, 32, lane);

  v8f cz = {};
  v8f o[4] = {cz, cz, cz, cz};
  float rm[8], rl[8];
#pragma unroll
  for (int r = 0; r < 8; ++r) { rm[r] = -1e30f; rl[r] = 0.0f; }

  for (int kv = 0; kv < S_; kv += 64) {
    // speculative prefetch of the next K/V tiles (-> global_prefetch_b8)
    if (kv + 64 < S_) {
      __builtin_prefetch(Kh + (size_t)(kv + 64 + lane) * DK_, 0, 1);
      __builtin_prefetch(Kh + (size_t)(kv + 96 + lane) * DK_, 0, 1);
      __builtin_prefetch(Vh + (size_t)(kv + 64 + lane) * DK_, 0, 1);
      __builtin_prefetch(Vh + (size_t)(kv + 96 + lane) * DK_, 0, 1);
    }
    __syncthreads();                                  // vt reuse guard
    // cooperative transposed V tile: vt[dk][key] <- Vh[kv+key][dk]
#pragma unroll
    for (int c8 = 0; c8 < 4; ++c8) {
      const int chunk = threadIdx.x + c8 * 128;       // 0..511
      const int key = chunk >> 3;
      const int dk0 = (chunk & 7) * 8;
      union { u32x4 q; f16 hh[8]; } tr;
      tr.q = *(const u32x4*)(Vh + (size_t)(kv + key) * DK_ + dk0);
#pragma unroll
      for (int i = 0; i < 8; ++i) vt[(dk0 + i) * 64 + key] = tr.hh[i];
    }
    __syncthreads();

    // scores S = Q * K^T  (16 x 64)
    v8f sv[4];
#pragma unroll
    for (int tj = 0; tj < 4; ++tj) {
      v16h bk0 = load_frag(Kh, DK_, kv + tj * 16, 0,  lane);
      v16h bk1 = load_frag(Kh, DK_, kv + tj * 16, 32, lane);
      sv[tj] = wmma16(aq1, bk1, wmma16(aq0, bk0, cz));
    }

    // key padding mask
#pragma unroll
    for (int tj = 0; tj < 4; ++tj) {
      if (mrow[kv + tj * 16 + col]) {
#pragma unroll
        for (int r = 0; r < 8; ++r) sv[tj][r] = NEG_;
      }
    }

    // online softmax (row lives across 16 lanes of one half)
#pragma unroll
    for (int r = 0; r < 8; ++r) {
      float v = fmaxf(fmaxf(sv[0][r], sv[1][r]), fmaxf(sv[2][r], sv[3][r]));
      v = fmaxf(v, __shfl_xor(v, 1));
      v = fmaxf(v, __shfl_xor(v, 2));
      v = fmaxf(v, __shfl_xor(v, 4));
      v = fmaxf(v, __shfl_xor(v, 8));
      const float mnew  = fmaxf(rm[r], v);
      const float alpha = __expf(rm[r] - mnew);
      float ls = 0.0f;
#pragma unroll
      for (int tj = 0; tj < 4; ++tj) {
        const float p = __expf(sv[tj][r] - mnew);
        sv[tj][r] = p;
        ls += p;
      }
      ls += __shfl_xor(ls, 1);
      ls += __shfl_xor(ls, 2);
      ls += __shfl_xor(ls, 4);
      ls += __shfl_xor(ls, 8);
      rl[r] = rl[r] * alpha + ls;
      rm[r] = mnew;
#pragma unroll
      for (int tj = 0; tj < 4; ++tj) o[tj][r] *= alpha;
    }

    // P (C-layout) -> LDS -> A-fragment layout
#pragma unroll
    for (int tj = 0; tj < 4; ++tj)
#pragma unroll
      for (int r = 0; r < 8; ++r)
        pw[(half * 8 + r) * 64 + tj * 16 + col] = (f16)sv[tj][r];

    v16h ap0 = load_frag(pw, 64, 0, 0,  lane);
    v16h ap1 = load_frag(pw, 64, 0, 32, lane);

    // O += P * V  (B-fragments from transposed V tile in LDS)
#pragma unroll
    for (int tj = 0; tj < 4; ++tj) {
      v16h bv0 = load_frag(vt, 64, tj * 16, 0,  lane);
      v16h bv1 = load_frag(vt, 64, tj * 16, 32, lane);
      o[tj] = wmma16(ap1, bv1, wmma16(ap0, bv0, o[tj]));
    }
  }

  // normalize + store ctx [B*S, D] with head-interleaved columns
#pragma unroll
  for (int tj = 0; tj < 4; ++tj) {
#pragma unroll
    for (int r = 0; r < 8; ++r) {
      const int srow = qrow0 + half * 8 + r;
      const float val = o[tj][r] / rl[r];
      ctx[((size_t)b * S_ + srow) * D_ + h * 64 + tj * 16 + col] = (f16)val;
    }
  }
}

// ---------------------------------------------------------------- launcher
extern "C" void kernel_launch(void* const* d_in, const int* in_sizes, int n_in,
                              void* d_out, int out_size, void* d_ws, size_t ws_size,
                              hipStream_t stream) {
  const float* x    = (const float*)d_in[0];
  const float* Wqkv = (const float*)d_in[1];
  const float* Wout = (const float*)d_in[2];
  const unsigned char* mask = (const unsigned char*)d_in[3];
  float* out = (float*)d_out;

  // workspace carve-up (f16 elements, ~48 MB total)
  f16* p = (f16*)d_ws;
  f16* xb    = p; p += (size_t)B_ * S_ * D_;     // 4 Mi
  f16* wqkvb = p; p += (size_t)E3_ * D_;         // 3 Mi
  f16* woutb = p; p += (size_t)D_ * D_;          // 1 Mi
  f16* Qb    = p; p += (size_t)B_ * H_ * S_ * DK_;
  f16* Kb    = p; p += (size_t)B_ * H_ * S_ * DK_;
  f16* Vb    = p; p += (size_t)B_ * H_ * S_ * DK_;
  f16* ctxb  = p;

  const int nx = B_ * S_ * D_;
  const int nw = E3_ * D_;
  const int no = D_ * D_;
  convert_f32_f16<<<(nx + 255) / 256, 256, 0, stream>>>(x, xb, nx);
  convert_f32_f16<<<(nw + 255) / 256, 256, 0, stream>>>(Wqkv, wqkvb, nw);
  convert_f32_f16<<<(no + 255) / 256, 256, 0, stream>>>(Wout, woutb, no);

  qkv_gemm<<<dim3(32, 24), 256, 0, stream>>>(xb, wqkvb, Qb, Kb, Vb);
  attention<<<dim3(32, 32), 128, 0, stream>>>(Qb, Kb, Vb, mask, ctxb);
  out_gemm<<<dim3(32, 8), 256, 0, stream>>>(ctxb, woutb, out);
}
